// EncoderLayer_88167088652684
// MI455X (gfx1250) — matrix operand order repaired and checked by reading, not compiled, not run on hardware
//
#include <hip/hip_runtime.h>

// ---------------------------------------------------------------------------
// CDNA5 (gfx1250) fused transformer encoder layer, M-blocked.
// Each workgroup (8 waves) processes BM=4 batch elements so every weight
// B-fragment loaded from L2 feeds 4 v_wmma_f32_16x16x32_bf16 ops (4x less
// L2 weight traffic than one-batch-per-block). All activations staged in
// LDS as bf16 (299 KB of the 320 KB WGP pool, 1 workgroup/WGP).
// ---------------------------------------------------------------------------

typedef __attribute__((ext_vector_type(16))) __bf16       v16bf;
typedef __attribute__((ext_vector_type(8)))  float        v8f;
typedef __attribute__((ext_vector_type(4)))  unsigned int u32x4;

#define LDX 520                  // padded row stride (elements)
#define ABS (16 * LDX)           // per-batch activation stride (elements)
#define BM  4                    // batch elements per block
#define SMEM_BYTES 299008        // 4*(XB,QB,KB,VB)=4*66560 + 4*ATT(8192)

// A-matrix 16x32 bf16 fragment (ISA 7.12.2): lane (m=lane&15, half=lane>>4)
// holds K = {k0+half*8 .. +7} then {k0+16+half*8 .. +7}.
__device__ __forceinline__ v16bf ldfragA(const __bf16* p) {
  union { v16bf v; u32x4 q[2]; } u;
  u.q[0] = *(const u32x4*)p;
  u.q[1] = *(const u32x4*)(p + 16);
  return u.v;
}
// B-matrix 32x16 bf16 fragment: lane holds column n=lane&15,
// K = {k0+half*8 .. +7} then {k0+16+half*8 .. +7}.
__device__ __forceinline__ v16bf ldfragB(const __bf16* p) {
  union { v16bf v; u32x4 q[2]; } u;
  u.q[0] = *(const u32x4*)p;
  u.q[1] = *(const u32x4*)(p + 16);
  return u.v;
}

template<int M, int T>
__device__ __forceinline__ void vzero(v8f (&a)[M][T]) {
#pragma unroll
  for (int i = 0; i < M; ++i)
#pragma unroll
    for (int j = 0; j < T; ++j)
#pragma unroll
      for (int r = 0; r < 8; ++r) a[i][j][r] = 0.0f;
}

// Multi-batch GEMM: C[bm] += A[bm](16xK) * W^T, W stored [N][K] row-major.
// One B-fragment load feeds M_ WMMAs (the whole point of M-blocking).
template<int TPW, int M_>
__device__ __forceinline__ void gemm_mb(
    const __bf16* Abase, int abstride, int lda, int akoff,
    const __bf16* W, int ldw, int wkoff,
    int K, int tile_base, int lane, v8f (&acc)[M_][TPW])
{
  const int m    = lane & 15;
  const int half = lane >> 4;
  for (int k0 = 0; k0 < K; k0 += 32) {
    v16bf af[M_];
#pragma unroll
    for (int bm = 0; bm < M_; ++bm)
      af[bm] = ldfragA(Abase + bm * abstride + m * lda + akoff + k0 + half * 8);
#pragma unroll
    for (int t = 0; t < TPW; ++t) {
      const int n = (tile_base + t) * 16 + m;
      v16bf bf = ldfragB(W + (size_t)n * ldw + wkoff + k0 + half * 8);
#pragma unroll
      for (int bm = 0; bm < M_; ++bm)
        acc[bm][t] = __builtin_amdgcn_wmma_f32_16x16x32_bf16(
            false, af[bm], false, bf, (short)0, acc[bm][t], false, false);
    }
  }
}

// ---------------------------------------------------------------------------
// Prep: fp32 [K][N] -> bf16 [N][K]; weights then live in the 192 MB L2.
// ---------------------------------------------------------------------------
__global__ void convT_kernel(const float* __restrict__ src,
                             __bf16* __restrict__ dst, int K, int N) {
  const size_t total = (size_t)K * N;
  for (size_t i = blockIdx.x * (size_t)blockDim.x + threadIdx.x; i < total;
       i += (size_t)gridDim.x * blockDim.x) {
    const size_t n = i / (size_t)K, k = i % (size_t)K;
    dst[i] = (__bf16)src[k * (size_t)N + n];
  }
}

__global__ void pe_kernel(float* __restrict__ pe) {
  int i = blockIdx.x * blockDim.x + threadIdx.x;
  if (i >= 16 * 512) return;
  int s = i >> 9, d = i & 511;
  float ang = (float)s * __expf((float)(d & ~1) * (-9.21034037198f / 512.0f));
  pe[i] = (d & 1) ? __cosf(ang) : __sinf(ang);
}

// ---------------------------------------------------------------------------
// LDS layout (bytes), per-batch stride 16640 within each region:
//   XB  @0       4 x [16][520] bf16  x, later y = LN1(...)
//   QB  @66560   4 x [16][520] bf16  q, later ctx, later FFN hidden quarter
//   KB  @133120  4 x [16][520] bf16  k
//   VB  @199680  4 x [16][520] bf16  v, later y0 / z staging for LayerNorm
//   ATT @266240  4 x [8][16][16] f32 attention probabilities
// ---------------------------------------------------------------------------
__global__ __launch_bounds__(256) void enc_layer(
    const float* __restrict__ enc, const float* __restrict__ pe,
    const __bf16* __restrict__ wqT, const __bf16* __restrict__ wkT,
    const __bf16* __restrict__ wvT, const __bf16* __restrict__ woT,
    const __bf16* __restrict__ w1T, const __bf16* __restrict__ w2T,
    const float* __restrict__ bq, const float* __restrict__ bk,
    const float* __restrict__ bv, const float* __restrict__ bo,
    const float* __restrict__ b1, const float* __restrict__ b2,
    const float* __restrict__ ln1g, const float* __restrict__ ln1b,
    const float* __restrict__ ln2g, const float* __restrict__ ln2b,
    const float* __restrict__ cw,
    float* __restrict__ out, float* __restrict__ attn_out)
{
  extern __shared__ __align__(16) char smem[];
  __bf16* XB  = (__bf16*)(smem);
  __bf16* QB  = (__bf16*)(smem + 66560);
  __bf16* KB  = (__bf16*)(smem + 133120);
  __bf16* VB  = (__bf16*)(smem + 199680);
  float*  ATT = (float*)(smem + 266240);   // + bm*2048 floats
  __bf16* HB  = QB;                        // FFN hidden quarter reuses QB

  const int tid  = threadIdx.x;
  const int wave = tid >> 5, lane = tid & 31;
  const int m    = lane & 15, half = lane >> 4;
  const size_t b0 = (size_t)blockIdx.x * BM;

  // 1. x = enc + positional encoding -> bf16 (also serves as residual 1)
  const float* encb = enc + b0 * 8192;
  for (int i = tid; i < BM * 8192; i += 256) {
    const int bm = i >> 13, j = i & 8191;
    const int s = j >> 9, d = j & 511;
    XB[bm * ABS + s * LDX + d] = (__bf16)(encb[i] + pe[j]);
  }
  __syncthreads();

  // 2. Q/K/V projections: 32 N-tiles, 4 per wave, x4 batches
  {
    v8f qa[BM][4]; vzero(qa);
    gemm_mb<4, BM>(XB, ABS, LDX, 0, wqT, 512, 0, 512, wave * 4, lane, qa);
#pragma unroll
    for (int bm = 0; bm < BM; ++bm)
#pragma unroll
      for (int t = 0; t < 4; ++t) {
        const int n = (wave * 4 + t) * 16 + m;
        const float bb = bq[n];
#pragma unroll
        for (int r = 0; r < 8; ++r)
          QB[bm * ABS + (r + 8 * half) * LDX + n] = (__bf16)(qa[bm][t][r] + bb);
      }
    v8f ka[BM][4]; vzero(ka);
    gemm_mb<4, BM>(XB, ABS, LDX, 0, wkT, 512, 0, 512, wave * 4, lane, ka);
#pragma unroll
    for (int bm = 0; bm < BM; ++bm)
#pragma unroll
      for (int t = 0; t < 4; ++t) {
        const int n = (wave * 4 + t) * 16 + m;
        const float bb = bk[n];
#pragma unroll
        for (int r = 0; r < 8; ++r)
          KB[bm * ABS + (r + 8 * half) * LDX + n] = (__bf16)(ka[bm][t][r] + bb);
      }
    v8f va[BM][4]; vzero(va);
    gemm_mb<4, BM>(XB, ABS, LDX, 0, wvT, 512, 0, 512, wave * 4, lane, va);
#pragma unroll
    for (int bm = 0; bm < BM; ++bm)
#pragma unroll
      for (int t = 0; t < 4; ++t) {
        const int n = (wave * 4 + t) * 16 + m;
        const float bb = bv[n];
#pragma unroll
        for (int r = 0; r < 8; ++r)
          VB[bm * ABS + (r + 8 * half) * LDX + n] = (__bf16)(va[bm][t][r] + bb);
      }
  }
  __syncthreads();

  // 3. scores = q.k^T / 8; wave == head, per-batch A and B
#pragma unroll
  for (int bm = 0; bm < BM; ++bm) {
    const __bf16* qb = QB + bm * ABS;
    const __bf16* kb = KB + bm * ABS;
    v8f sc;
#pragma unroll
    for (int r = 0; r < 8; ++r) sc[r] = 0.0f;
#pragma unroll
    for (int k0 = 0; k0 < 64; k0 += 32) {
      v16bf af  = ldfragA(qb + m * LDX + wave * 64 + k0 + half * 8);
      v16bf bfr = ldfragB(kb + m * LDX + wave * 64 + k0 + half * 8);
      sc = __builtin_amdgcn_wmma_f32_16x16x32_bf16(
          false, af, false, bfr, (short)0, sc, false, false);
    }
#pragma unroll
    for (int r = 0; r < 8; ++r)
      ATT[bm * 2048 + (wave * 16 + r + 8 * half) * 16 + m] = sc[r] * 0.125f;
  }
  __syncthreads();

  // 4. softmax over key dim: 512 rows (4 batches x 8 heads x 16 q), 2/thread
#pragma unroll
  for (int rr = 0; rr < 2; ++rr) {
    const int r = tid + rr * 256;
    float* row = ATT + (r >> 7) * 2048 + (r & 127) * 16;
    float mx = row[0];
#pragma unroll
    for (int j = 1; j < 16; ++j) mx = fmaxf(mx, row[j]);
    float sum = 0.f;
#pragma unroll
    for (int j = 0; j < 16; ++j) { float e = __expf(row[j] - mx); row[j] = e; sum += e; }
    const float inv = 1.f / sum;
#pragma unroll
    for (int j = 0; j < 16; ++j) row[j] *= inv;
  }
  __syncthreads();

  // 5. attn_out[b,s,t] = sum_h cw[h] * attn[h,s,t]   (4*256 values)
#pragma unroll
  for (int ii = 0; ii < BM; ++ii) {
    const int i = tid + ii * 256;
    const int bm = i >> 8, s = (i >> 4) & 15, t = i & 15;
    float a = 0.f;
#pragma unroll
    for (int h = 0; h < 8; ++h) a += cw[h] * ATT[bm * 2048 + (h * 16 + s) * 16 + t];
    attn_out[(b0 + bm) * 256 + (s * 16 + t)] = a;
  }

  // 6. ctx = attn @ v (K=16, <1% of FLOPs -> VALU); overwrite QB
  for (int i = tid; i < BM * 8192; i += 256) {
    const int bm = i >> 13, j = i & 8191;
    const int s = j >> 9, d = j & 511, h = d >> 6;
    const float* ar = ATT + bm * 2048 + (h * 16 + s) * 16;
    float a = 0.f;
#pragma unroll
    for (int t = 0; t < 16; ++t) a += ar[t] * (float)VB[bm * ABS + t * LDX + d];
    QB[bm * ABS + s * LDX + d] = (__bf16)a;
  }
  __syncthreads();

  // 7. y0 = ctx@wo + bo + x -> VB (bf16 staging for LN1)
  {
    v8f oa[BM][4]; vzero(oa);
    gemm_mb<4, BM>(QB, ABS, LDX, 0, woT, 512, 0, 512, wave * 4, lane, oa);
    __syncthreads();                       // VB (v) fully consumed by step 6
#pragma unroll
    for (int bm = 0; bm < BM; ++bm)
#pragma unroll
      for (int t = 0; t < 4; ++t) {
        const int n = (wave * 4 + t) * 16 + m;
        const float bb = bo[n];
#pragma unroll
        for (int r = 0; r < 8; ++r) {
          const int row = r + 8 * half;
          VB[bm * ABS + row * LDX + n] =
              (__bf16)(oa[bm][t][r] + bb + (float)XB[bm * ABS + row * LDX + n]);
        }
      }
  }
  __syncthreads();

  // 8. LN1(y0) -> XB (y, bf16): 64 rows, 8 per wave, wave32 shfl reductions
#pragma unroll
  for (int rr = 0; rr < 8; ++rr) {
    const int r = wave * 8 + rr;
    const int bm = r >> 4, s = r & 15;
    const __bf16* src = VB + bm * ABS + s * LDX;
    float sum = 0.f, sq = 0.f;
    for (int d = lane; d < 512; d += 32) { float v = (float)src[d]; sum += v; sq += v * v; }
#pragma unroll
    for (int off = 16; off > 0; off >>= 1) { sum += __shfl_xor(sum, off); sq += __shfl_xor(sq, off); }
    const float mu = sum * (1.f / 512.f);
    const float rs = rsqrtf(sq * (1.f / 512.f) - mu * mu + 1e-5f);
    for (int d = lane; d < 512; d += 32)
      XB[bm * ABS + s * LDX + d] =
          (__bf16)(((float)src[d] - mu) * rs * ln1g[d] + ln1b[d]);
  }
  __syncthreads();

  // 9. FFN, K-split into quarters of F=2048: h=relu(y@w1+b1) into HB (=QB),
  //    then out += h@w2 accumulated in persistent registers.
  v8f oacc[BM][4]; vzero(oacc);
#pragma unroll 1
  for (int qf = 0; qf < 4; ++qf) {
    v8f ha[BM][4]; vzero(ha);
    gemm_mb<4, BM>(XB, ABS, LDX, 0, w1T + (size_t)qf * 512 * 512, 512, 0, 512,
                   wave * 4, lane, ha);
    __syncthreads();                       // prior-quarter HB reads complete
#pragma unroll
    for (int bm = 0; bm < BM; ++bm)
#pragma unroll
      for (int t = 0; t < 4; ++t) {
        const int nl = (wave * 4 + t) * 16 + m;
        const float bb = b1[qf * 512 + nl];
#pragma unroll
        for (int r = 0; r < 8; ++r)
          HB[bm * ABS + (r + 8 * half) * LDX + nl] =
              (__bf16)fmaxf(ha[bm][t][r] + bb, 0.f);
      }
    __syncthreads();
    gemm_mb<4, BM>(HB, ABS, LDX, 0, w2T, 2048, qf * 512, 512, wave * 4, lane, oacc);
  }
  __syncthreads();

  // 10. z = ffn + b2 + y -> VB; LN2 -> global out
#pragma unroll
  for (int bm = 0; bm < BM; ++bm)
#pragma unroll
    for (int t = 0; t < 4; ++t) {
      const int n = (wave * 4 + t) * 16 + m;
      const float bb = b2[n];
#pragma unroll
      for (int r = 0; r < 8; ++r) {
        const int row = r + 8 * half;
        VB[bm * ABS + row * LDX + n] =
            (__bf16)(oacc[bm][t][r] + bb + (float)XB[bm * ABS + row * LDX + n]);
      }
    }
  __syncthreads();
#pragma unroll
  for (int rr = 0; rr < 8; ++rr) {
    const int r = wave * 8 + rr;
    const int bm = r >> 4, s = r & 15;
    const __bf16* src = VB + bm * ABS + s * LDX;
    float sum = 0.f, sq = 0.f;
    for (int d = lane; d < 512; d += 32) { float v = (float)src[d]; sum += v; sq += v * v; }
#pragma unroll
    for (int off = 16; off > 0; off >>= 1) { sum += __shfl_xor(sum, off); sq += __shfl_xor(sq, off); }
    const float mu = sum * (1.f / 512.f);
    const float rs = rsqrtf(sq * (1.f / 512.f) - mu * mu + 1e-5f);
    for (int d = lane; d < 512; d += 32)
      out[(b0 + bm) * 8192 + s * 512 + d] =
          ((float)src[d] - mu) * rs * ln2g[d] + ln2b[d];
  }
}

// ---------------------------------------------------------------------------
extern "C" void kernel_launch(void* const* d_in, const int* in_sizes, int n_in,
                              void* d_out, int out_size, void* d_ws, size_t ws_size,
                              hipStream_t stream) {
  const float* enc  = (const float*)d_in[0];
  const float* wq   = (const float*)d_in[1];
  const float* bq   = (const float*)d_in[2];
  const float* wk   = (const float*)d_in[3];
  const float* bk   = (const float*)d_in[4];
  const float* wv   = (const float*)d_in[5];
  const float* bv   = (const float*)d_in[6];
  const float* wo   = (const float*)d_in[7];
  const float* bo   = (const float*)d_in[8];
  const float* ln1g = (const float*)d_in[9];
  const float* ln1b = (const float*)d_in[10];
  const float* w1   = (const float*)d_in[11];
  const float* b1   = (const float*)d_in[12];
  const float* w2   = (const float*)d_in[13];
  const float* b2   = (const float*)d_in[14];
  const float* ln2g = (const float*)d_in[15];
  const float* ln2b = (const float*)d_in[16];
  const float* cw   = (const float*)d_in[17];

  // workspace: transposed bf16 weights + positional encoding (~6.03 MB)
  __bf16* wqT = (__bf16*)d_ws;
  __bf16* wkT = wqT + 262144;
  __bf16* wvT = wkT + 262144;
  __bf16* woT = wvT + 262144;
  __bf16* w1T = woT + 262144;            // [2048][512]
  __bf16* w2T = w1T + 1048576;           // [512][2048]
  float*  pe  = (float*)((char*)d_ws + 6291456);

  convT_kernel<<<512, 256, 0, stream>>>(wq, wqT, 512, 512);
  convT_kernel<<<512, 256, 0, stream>>>(wk, wkT, 512, 512);
  convT_kernel<<<512, 256, 0, stream>>>(wv, wvT, 512, 512);
  convT_kernel<<<512, 256, 0, stream>>>(wo, woT, 512, 512);
  convT_kernel<<<1024, 256, 0, stream>>>(w1, w1T, 512, 2048);
  convT_kernel<<<1024, 256, 0, stream>>>(w2, w2T, 2048, 512);
  pe_kernel<<<32, 256, 0, stream>>>(pe);

  const int B = in_sizes[0] / 8192;      // 4096
  float* outp     = (float*)d_out;
  float* attn_out = outp + (size_t)B * 8192;

  enc_layer<<<B / BM, 256, SMEM_BYTES, stream>>>(
      enc, pe, wqT, wkT, wvT, woT, w1T, w2T,
      bq, bk, bv, bo, b1, b2, ln1g, ln1b, ln2g, ln2b, cw,
      outp, attn_out);
}
